// DiffusionTransformerBlock_25305947308313
// MI455X (gfx1250) — matrix operand order repaired and checked by reading, not compiled, not run on hardware
//
#include <hip/hip_runtime.h>

// ---------------------------------------------------------------------------
// Types for CDNA5 WMMA (wave32, 16x16x32 bf16 -> f32) and TDM descriptors
// ---------------------------------------------------------------------------
typedef __attribute__((ext_vector_type(16))) __bf16          v16bf;
typedef __attribute__((ext_vector_type(8)))  float           v8f;
typedef __attribute__((ext_vector_type(8)))  unsigned short  ushort8;
typedef __attribute__((ext_vector_type(4)))  unsigned int    uint4v;
typedef __attribute__((ext_vector_type(8)))  int             int8v;
typedef __attribute__((ext_vector_type(4)))  int             int4v;

#if defined(__HIP_DEVICE_COMPILE__) && __has_builtin(__builtin_amdgcn_tensor_load_to_lds)
#define HAVE_TDM 1
#else
#define HAVE_TDM 0
#endif

union FragU  { v16bf v; ushort8 h[2]; };            // fragment from 2x b128 loads
union FragW  { v16bf v; unsigned short u[16]; };    // fragment built element-wise
union AccU   { v8f v; float f[8]; };

struct __align__(8) US4 { unsigned short x, y, z, w; };

__device__ __forceinline__ unsigned short f2bf(float f) {
  union { float f; unsigned u; } x; x.f = f;
  unsigned r = x.u + 0x7FFFu + ((x.u >> 16) & 1u);  // round-to-nearest-even
  return (unsigned short)(r >> 16);
}
__device__ __forceinline__ float sigmoidf(float x) { return 1.0f / (1.0f + __expf(-x)); }

// ---------------------------------------------------------------------------
// WMMA K-loop over J adjacent 16x16 output tiles sharing one A fragment.
// A [16 x Kd] rows at stride lda (ushorts, bf16); BT rows = B^T at stride ldb.
// Lane layouts per CDNA5 ISA 7.12.2 (16-bit A 16x32 / B 32x16).
// ---------------------------------------------------------------------------
template <int J>
__device__ __forceinline__ void wmma_tiles(const unsigned short* A, int lda,
                                           const unsigned short* BT, int ldb,
                                           int Kd, int lane, v8f* acc) {
  const int half = lane >> 4;
  const int l15  = lane & 15;
  const unsigned short* arow = A + (size_t)l15 * lda;
  const int kA = 8 * half;
  const int kB = 8 * half;
  for (int kk = 0; kk < Kd; kk += 32) {
    if (kk + 64 <= Kd) __builtin_prefetch(arow + kk + 64, 0, 3);
    FragU fa;
    fa.h[0] = *(const ushort8*)(arow + kk + kA);
    fa.h[1] = *(const ushort8*)(arow + kk + kA + 16);
#pragma unroll
    for (int j = 0; j < J; ++j) {
      const unsigned short* brow = BT + (size_t)(j * 16 + l15) * ldb;
      if (kk + 64 <= Kd) __builtin_prefetch(brow + kk + 64, 0, 3);
      FragU fb;
      fb.h[0] = *(const ushort8*)(brow + kk + kB);
      fb.h[1] = *(const ushort8*)(brow + kk + kB + 16);
      acc[j] = __builtin_amdgcn_wmma_f32_16x16x32_bf16(false, fa.v, false, fb.v,
                                                       (short)0, acc[j], false, false);
    }
  }
}

// ---------------------------------------------------------------------------
// Generic GEMM: C[M,Nc](f32, ldc) = A[M,Kd](bf16) @ BT[Nc,Kd]^T + bias
// one wave per 16x64 output strip; grid = (Nc/64, M/16)
// ---------------------------------------------------------------------------
__global__ void gemm_kernel(const unsigned short* __restrict__ A, int lda,
                            const unsigned short* __restrict__ BT, int ldb,
                            const float* __restrict__ bias,
                            float* __restrict__ C, int ldc, int Kd) {
  const int lane = threadIdx.x;
  const int tn = blockIdx.x, tm = blockIdx.y;
  v8f acc[4] = {};
  wmma_tiles<4>(A + (size_t)tm * 16 * lda, lda,
                BT + (size_t)tn * 64 * ldb, ldb, Kd, lane, acc);
  const int mbase = tm * 16 + 8 * (lane >> 4);
#pragma unroll
  for (int j = 0; j < 4; ++j) {
    const int col = tn * 64 + j * 16 + (lane & 15);
    const float b = bias ? bias[col] : 0.0f;
    AccU u; u.v = acc[j];
    for (int r = 0; r < 8; ++r)
      C[(size_t)(mbase + r) * ldc + col] = u.f[r] + b;
  }
}

// ---------------------------------------------------------------------------
// z path (HBM-bound stage): per block, 64 contiguous (q,k) rows of
// z[1024,1024,128] staged into LDS by the Tensor Data Mover (fallback:
// cooperative loads), LayerNorm*gamma_z, then (16x128)@WzT(16x128) via
// 4 WMMA K-steps per wave -> bias_ws[h][q][k].  Block = 128 thr (4 waves).
// ---------------------------------------------------------------------------
__global__ void zbias_kernel(const float* __restrict__ z,
                             const float* __restrict__ gz,
                             const unsigned short* __restrict__ WzT,
                             float* __restrict__ bias_ws) {
  __shared__ float tile[64 * 128];                   // 32 KB z tile
  __shared__ float mu[64], rs[64];
  __shared__ float ps[128], pq[128];
  const int tid  = threadIdx.x;
  const int lane = tid & 31;
  const int wv   = tid >> 5;
  const size_t t = blockIdx.x;                       // 64-row tile over q*N+k
  const float* src = z + t * (64 * 128);

#if HAVE_TDM
  if (wv == 0) {                                     // one TDM issue per block
    const unsigned TD0 = 64u * 128u;                 // 8192 elements, contiguous
    unsigned lds_base = (unsigned)(uintptr_t)(void*)tile;
    unsigned long long ga = (unsigned long long)(uintptr_t)src;
    uint4v g0;
    g0.x = 1u;                                       // D# count = 1
    g0.y = lds_base;                                 // lds_addr
    g0.z = (unsigned)(ga & 0xFFFFFFFFu);             // global_addr[31:0]
    g0.w = (unsigned)((ga >> 32) & 0x01FFFFFFu) | (2u << 30);  // addr[56:32] | type=2
    int8v g1;
    g1[0] = 0x00020000;                              // data_size = 4B
    g1[1] = (int)(TD0 << 16);                        // tensor_dim0[15:0]
    g1[2] = (int)((TD0 >> 16) | (1u << 16));         // tensor_dim0[31:16], tensor_dim1=1
    g1[3] = (int)(TD0 << 16);                        // tile_dim0 = 8192
    g1[4] = 1;                                       // tile_dim1 = 1
    g1[5] = (int)TD0;                                // tensor_dim0_stride
    g1[6] = 0; g1[7] = 0;
    int4v gnull = {0, 0, 0, 0};
#if __clang_major__ >= 23
    int8v g1b = {};
    __builtin_amdgcn_tensor_load_to_lds(g0, g1, gnull, gnull, g1b, 0);
#else
    __builtin_amdgcn_tensor_load_to_lds(g0, g1, gnull, gnull, 0);
#endif
    __builtin_amdgcn_s_wait_tensorcnt(0);
  }
#else
  for (int i = 0; i < 16; ++i) {                     // 32 KB cooperative load
    int idx = (i * 128 + tid) * 4;
    *(float4*)(tile + idx) = *(const float4*)(src + idx);
  }
#endif
  __syncthreads();

  // LN statistics: 2 lanes per row, 64 channels each
  {
    int lr = wv * 16 + (lane & 15);
    int c0 = (lane >> 4) * 64;
    float s = 0.f, ss = 0.f;
    for (int c = 0; c < 64; ++c) { float x = tile[lr * 128 + c0 + c]; s += x; ss += x * x; }
    ps[tid] = s; pq[tid] = ss;
  }
  __syncthreads();
  if (lane < 16) {
    int lr = wv * 16 + lane;
    float s = ps[wv * 32 + lane] + ps[wv * 32 + 16 + lane];
    float q = pq[wv * 32 + lane] + pq[wv * 32 + 16 + lane];
    float m = s * (1.0f / 128.0f);
    float v = q * (1.0f / 128.0f) - m * m;
    mu[lr] = m; rs[lr] = rsqrtf(v + 1e-5f);
  }
  __syncthreads();

  const int half = lane >> 4;
  const int mrow = wv * 16 + (lane & 15);            // local row for A fragment
  const int kA = 8 * half, kB = 8 * half;
  const float m = mu[mrow], r = rs[mrow];
  const unsigned short* brow = WzT + (lane & 15) * 128;
  v8f acc = {};
  for (int kk = 0; kk < 128; kk += 32) {
    FragW fa;
#pragma unroll
    for (int e = 0; e < 8; ++e) {
      int k0 = kk + kA + e;
      int k1 = kk + kA + 16 + e;
      fa.u[e]     = f2bf((tile[mrow * 128 + k0] - m) * r * gz[k0]);
      fa.u[e + 8] = f2bf((tile[mrow * 128 + k1] - m) * r * gz[k1]);
    }
    FragU fb;
    fb.h[0] = *(const ushort8*)(brow + kk + kB);
    fb.h[1] = *(const ushort8*)(brow + kk + kB + 16);
    acc = __builtin_amdgcn_wmma_f32_16x16x32_bf16(false, fa.v, false, fb.v,
                                                  (short)0, acc, false, false);
  }
  AccU u; u.v = acc;
  const size_t h = (size_t)(lane & 15);
  for (int rr = 0; rr < 8; ++rr) {
    size_t gr = t * 64 + (size_t)(wv * 16 + rr + 8 * half);  // flattened q*1024+k
    bias_ws[(h << 20) + gr] = u.f[rr];
  }
}

// ---------------------------------------------------------------------------
// Attention scores: per (h, qtile, 64-wide k strip): Qb[h]@Kb[h]^T*scale+bias
// Qb/Kb: bf16 [H][1024][64] (D=48 zero-padded to 64)
// ---------------------------------------------------------------------------
__global__ void scores_kernel(const unsigned short* __restrict__ Qb,
                              const unsigned short* __restrict__ Kb,
                              const float* __restrict__ bias_ws,
                              float* __restrict__ scores) {
  const int lane = threadIdx.x;
  const int tkg = blockIdx.x, tq = blockIdx.y, h = blockIdx.z;
  const unsigned short* A = Qb + ((size_t)h * 1024 + tq * 16) * 64;
  const unsigned short* B = Kb + ((size_t)h * 1024 + tkg * 64) * 64;
  v8f acc[4] = {};
  wmma_tiles<4>(A, 64, B, 64, 64, lane, acc);
  const float scale = 0.14433756729740643f;          // 1/sqrt(48)
  const size_t base = ((size_t)h << 20);
  const int qbase = tq * 16 + 8 * (lane >> 4);
#pragma unroll
  for (int j = 0; j < 4; ++j) {
    const int kcol = tkg * 64 + j * 16 + (lane & 15);
    AccU u; u.v = acc[j];
    for (int r = 0; r < 8; ++r) {
      size_t idx = base + ((size_t)(qbase + r) << 10) + kcol;
      scores[idx] = u.f[r] * scale + bias_ws[idx];
    }
  }
}

// ---------------------------------------------------------------------------
// Row softmax over 1024 scores, written back IN PLACE as bf16 probs
// (first 2KB of each 4KB f32 row; PV kernel reads with lda=2048 ushorts).
// ---------------------------------------------------------------------------
__global__ void softmax_kernel(float* __restrict__ scores) {
  __shared__ float sred[256];
  const size_t row = blockIdx.x;                     // h*1024 + q
  float* p = scores + (row << 10);
  const int t = threadIdx.x;
  float4 v = *(const float4*)(p + t * 4);
  float m = fmaxf(fmaxf(v.x, v.y), fmaxf(v.z, v.w));
  sred[t] = m; __syncthreads();
  for (int s = 128; s > 0; s >>= 1) { if (t < s) sred[t] = fmaxf(sred[t], sred[t + s]); __syncthreads(); }
  m = sred[0]; __syncthreads();
  float4 e;
  e.x = __expf(v.x - m); e.y = __expf(v.y - m); e.z = __expf(v.z - m); e.w = __expf(v.w - m);
  sred[t] = e.x + e.y + e.z + e.w; __syncthreads();
  for (int s = 128; s > 0; s >>= 1) { if (t < s) sred[t] += sred[t + s]; __syncthreads(); }
  const float inv = 1.0f / sred[0];
  __syncthreads();                                   // all row reads complete
  unsigned short* o = (unsigned short*)p;
  US4 w; w.x = f2bf(e.x * inv); w.y = f2bf(e.y * inv); w.z = f2bf(e.z * inv); w.w = f2bf(e.w * inv);
  *(US4*)(o + t * 4) = w;
}

// ---------------------------------------------------------------------------
// P @ V: probs bf16 (in-place rows, lda=2048 ushorts) x VbT[h][48][1024]
// -> OH f32 [1024][768] at [q][h*48+d];  3 d-tiles per wave
// ---------------------------------------------------------------------------
__global__ void pv_kernel(const float* __restrict__ scores,
                          const unsigned short* __restrict__ VbT,
                          float* __restrict__ OH) {
  const int lane = threadIdx.x;
  const int tq = blockIdx.y, h = blockIdx.z;
  const unsigned short* A = (const unsigned short*)(scores + ((size_t)h << 20))
                            + (size_t)tq * 16 * 2048;
  const unsigned short* B = VbT + (size_t)h * 48 * 1024;
  v8f acc[3] = {};
  wmma_tiles<3>(A, 2048, B, 1024, 1024, lane, acc);
  const int qbase = tq * 16 + 8 * (lane >> 4);
#pragma unroll
  for (int j = 0; j < 3; ++j) {
    const int d = j * 16 + (lane & 15);
    AccU u; u.v = acc[j];
    for (int r = 0; r < 8; ++r)
      OH[(size_t)(qbase + r) * 768 + h * 48 + d] = u.f[r];
  }
}

// ---------------------------------------------------------------------------
// LayerNorm over last dim C: optional gamma, optional f32 / bf16 outputs
// ---------------------------------------------------------------------------
__global__ void ln_kernel(const float* __restrict__ x, const float* __restrict__ gamma,
                          float* __restrict__ of32, unsigned short* __restrict__ obf16,
                          int C) {
  __shared__ float sred[256];
  const size_t row = blockIdx.x;
  const float* xr = x + row * (size_t)C;
  const int t = threadIdx.x;
  float s = 0.f, ss = 0.f;
  for (int c = t; c < C; c += 256) { float v = xr[c]; s += v; ss += v * v; }
  sred[t] = s; __syncthreads();
  for (int k = 128; k > 0; k >>= 1) { if (t < k) sred[t] += sred[t + k]; __syncthreads(); }
  s = sred[0]; __syncthreads();
  sred[t] = ss; __syncthreads();
  for (int k = 128; k > 0; k >>= 1) { if (t < k) sred[t] += sred[t + k]; __syncthreads(); }
  ss = sred[0];
  const float mean = s / C;
  const float r = rsqrtf(ss / C - mean * mean + 1e-5f);
  for (int c = t; c < C; c += 256) {
    float y = (xr[c] - mean) * r;
    if (gamma) y *= gamma[c];
    if (of32)  of32[row * (size_t)C + c] = y;
    if (obf16) obf16[row * (size_t)C + c] = f2bf(y);
  }
}

// ---------------------------------------------------------------------------
// Elementwise / packing helpers
// ---------------------------------------------------------------------------
__global__ void transpose_bf16_kernel(const float* __restrict__ W,
                                      unsigned short* __restrict__ BT, int K, int Nc) {
  int i = blockIdx.x * 256 + threadIdx.x;
  if (i >= K * Nc) return;
  int k = i / Nc, n = i % Nc;
  BT[(size_t)n * K + k] = f2bf(W[i]);
}
__global__ void conv_bf16_kernel(const float* __restrict__ x,
                                 unsigned short* __restrict__ o, int n) {
  int i = blockIdx.x * 256 + threadIdx.x;
  if (i < n) o[i] = f2bf(x[i]);
}
__global__ void ew_adaln_kernel(const float* __restrict__ G, const float* __restrict__ A,
                                const float* __restrict__ aln,
                                unsigned short* __restrict__ o, int n) {
  int i = blockIdx.x * 256 + threadIdx.x;
  if (i < n) o[i] = f2bf(sigmoidf(G[i]) * aln[i] + A[i]);
}
__global__ void ew_gatemul_kernel(const float* __restrict__ X, const float* __restrict__ Y,
                                  unsigned short* __restrict__ o, int n) {
  int i = blockIdx.x * 256 + threadIdx.x;
  if (i < n) o[i] = f2bf(sigmoidf(X[i]) * Y[i]);
}
__global__ void ew_silumul_kernel(const float* __restrict__ X1, const float* __restrict__ X2,
                                  unsigned short* __restrict__ o, int n) {
  int i = blockIdx.x * 256 + threadIdx.x;
  if (i < n) { float x = X1[i]; o[i] = f2bf(x * sigmoidf(x) * X2[i]); }
}
__global__ void final_kernel(const float* __restrict__ SL, const float* __restrict__ OO,
                             const float* __restrict__ ST, const float* __restrict__ TO,
                             float* __restrict__ out, int n) {
  int i = blockIdx.x * 256 + threadIdx.x;
  if (i < n) out[i] = sigmoidf(SL[i]) * OO[i] + sigmoidf(ST[i]) * TO[i];
}
// Q/K pack: f32 [1024][768] -> bf16 [16][1024][64] (d 48..63 zero)
__global__ void pack_qk_kernel(const float* __restrict__ X, unsigned short* __restrict__ o) {
  int i = blockIdx.x * 256 + threadIdx.x;
  if (i >= 16 * 1024 * 64) return;
  int d = i & 63, n = (i >> 6) & 1023, h = i >> 16;
  o[i] = (d < 48) ? f2bf(X[(size_t)n * 768 + h * 48 + d]) : (unsigned short)0;
}
// V pack transposed: f32 [1024][768] -> bf16 [16][48][1024]
__global__ void pack_vt_kernel(const float* __restrict__ V, unsigned short* __restrict__ o) {
  int i = blockIdx.x * 256 + threadIdx.x;
  if (i >= 16 * 48 * 1024) return;
  int n = i & 1023, d = (i >> 10) % 48, h = i / (48 * 1024);
  o[i] = f2bf(V[(size_t)n * 768 + h * 48 + d]);
}

// ---------------------------------------------------------------------------
// Host launcher
// ---------------------------------------------------------------------------
extern "C" void kernel_launch(void* const* d_in, const int* in_sizes, int n_in,
                              void* d_out, int out_size, void* d_ws, size_t ws_size,
                              hipStream_t stream) {
  (void)in_sizes; (void)n_in; (void)out_size; (void)ws_size;
  const int N = 1024, CA = 768, CS = 384, CZ = 128, H = 16, DP = 64;

  const float* a        = (const float*)d_in[0];
  const float* s        = (const float*)d_in[1];
  const float* z        = (const float*)d_in[2];
  const float* gamma_s1 = (const float*)d_in[3];
  const float* Wg1      = (const float*)d_in[4];
  const float* bg1      = (const float*)d_in[5];
  const float* Wadd1    = (const float*)d_in[6];
  const float* gamma_z  = (const float*)d_in[7];
  const float* Wz       = (const float*)d_in[8];
  const float* Wq       = (const float*)d_in[9];
  const float* bq       = (const float*)d_in[10];
  const float* Wk       = (const float*)d_in[11];
  const float* Wv       = (const float*)d_in[12];
  const float* Wgate    = (const float*)d_in[13];
  const float* bgate    = (const float*)d_in[14];
  const float* Wo       = (const float*)d_in[15];
  const float* bo       = (const float*)d_in[16];
  const float* Ws_last  = (const float*)d_in[17];
  const float* bs_last  = (const float*)d_in[18];
  const float* gamma_s2 = (const float*)d_in[19];
  const float* Wg2      = (const float*)d_in[20];
  const float* bg2      = (const float*)d_in[21];
  const float* Wadd2    = (const float*)d_in[22];
  const float* W1       = (const float*)d_in[23];
  const float* W2       = (const float*)d_in[24];
  const float* Wb       = (const float*)d_in[25];
  const float* Ws_t     = (const float*)d_in[26];
  const float* b_t      = (const float*)d_in[27];

  char* wp = (char*)d_ws;
  auto alloc = [&](size_t bytes) -> void* {
    void* p = (void*)wp; wp += (bytes + 255) & ~(size_t)255; return p;
  };
  // bf16 transposed weights
  unsigned short* WzT   = (unsigned short*)alloc((size_t)H * CZ * 2);
  unsigned short* Wg1T  = (unsigned short*)alloc((size_t)CA * CS * 2);
  unsigned short* Wa1T  = (unsigned short*)alloc((size_t)CA * CS * 2);
  unsigned short* WqT   = (unsigned short*)alloc((size_t)CA * CA * 2);
  unsigned short* WkT   = (unsigned short*)alloc((size_t)CA * CA * 2);
  unsigned short* WvT   = (unsigned short*)alloc((size_t)CA * CA * 2);
  unsigned short* WgT   = (unsigned short*)alloc((size_t)CA * CA * 2);
  unsigned short* WoT   = (unsigned short*)alloc((size_t)CA * CA * 2);
  unsigned short* WslT  = (unsigned short*)alloc((size_t)CA * CS * 2);
  unsigned short* Wg2T  = (unsigned short*)alloc((size_t)CA * CS * 2);
  unsigned short* Wa2T  = (unsigned short*)alloc((size_t)CA * CS * 2);
  unsigned short* WstT  = (unsigned short*)alloc((size_t)CA * CS * 2);
  unsigned short* W1T   = (unsigned short*)alloc((size_t)2 * CA * CA * 2);
  unsigned short* W2T   = (unsigned short*)alloc((size_t)2 * CA * CA * 2);
  unsigned short* WbT   = (unsigned short*)alloc((size_t)2 * CA * CA * 2);
  // activations
  unsigned short* sb    = (unsigned short*)alloc((size_t)N * CS * 2);
  float*          ALN   = (float*)alloc((size_t)N * CA * 4);
  unsigned short* SLN1  = (unsigned short*)alloc((size_t)N * CS * 2);
  unsigned short* SLN2  = (unsigned short*)alloc((size_t)N * CS * 2);
  float*          BIAS  = (float*)alloc((size_t)H * N * N * 4);   // 64 MB
  float*          SCORE = (float*)alloc((size_t)H * N * N * 4);   // 64 MB (probs in-place)
  float*          G1    = (float*)alloc((size_t)N * CA * 4);
  float*          ADD1  = (float*)alloc((size_t)N * CA * 4);
  unsigned short* ALN1b = (unsigned short*)alloc((size_t)N * CA * 2);
  float*          Qf    = (float*)alloc((size_t)N * CA * 4);
  float*          Kf    = (float*)alloc((size_t)N * CA * 4);
  float*          Vf    = (float*)alloc((size_t)N * CA * 4);
  float*          GATEf = (float*)alloc((size_t)N * CA * 4);
  unsigned short* Qb    = (unsigned short*)alloc((size_t)H * N * DP * 2);
  unsigned short* Kb    = (unsigned short*)alloc((size_t)H * N * DP * 2);
  unsigned short* VbT   = (unsigned short*)alloc((size_t)H * 48 * N * 2);
  float*          OH    = (float*)alloc((size_t)N * CA * 4);
  unsigned short* OGb   = (unsigned short*)alloc((size_t)N * CA * 2);
  float*          OOUT  = (float*)alloc((size_t)N * CA * 4);
  float*          SLAST = (float*)alloc((size_t)N * CA * 4);
  unsigned short* A2b   = (unsigned short*)alloc((size_t)N * CA * 2);
  float*          X1    = (float*)alloc((size_t)N * 2 * CA * 4);
  float*          X2    = (float*)alloc((size_t)N * 2 * CA * 4);
  unsigned short* BMIDb = (unsigned short*)alloc((size_t)N * 2 * CA * 2);
  float*          TOUT  = (float*)alloc((size_t)N * CA * 4);
  float*          ST    = (float*)alloc((size_t)N * CA * 4);

  auto tr = [&](const float* W, unsigned short* BT, int K, int Nc) {
    int n = K * Nc;
    transpose_bf16_kernel<<<(n + 255) / 256, 256, 0, stream>>>(W, BT, K, Nc);
  };
  auto gemm = [&](const unsigned short* A, int lda, const unsigned short* BT, int ldb,
                  const float* bias, float* C, int ldc, int M, int Nc, int Kd) {
    dim3 grid(Nc / 64, M / 16);
    gemm_kernel<<<grid, 32, 0, stream>>>(A, lda, BT, ldb, bias, C, ldc, Kd);
  };

  // --- weight prep (bf16, transposed to [Nc][K]) ---
  tr(Wz, WzT, CZ, H);
  tr(Wg1, Wg1T, CS, CA);   tr(Wadd1, Wa1T, CS, CA);
  tr(Wq, WqT, CA, CA);     tr(Wk, WkT, CA, CA);     tr(Wv, WvT, CA, CA);
  tr(Wgate, WgT, CA, CA);  tr(Wo, WoT, CA, CA);
  tr(Ws_last, WslT, CS, CA);
  tr(Wg2, Wg2T, CS, CA);   tr(Wadd2, Wa2T, CS, CA); tr(Ws_t, WstT, CS, CA);
  tr(W1, W1T, CA, 2 * CA); tr(W2, W2T, CA, 2 * CA); tr(Wb, WbT, 2 * CA, CA);
  conv_bf16_kernel<<<(N * CS + 255) / 256, 256, 0, stream>>>(s, sb, N * CS);

  // --- LayerNorms ---
  ln_kernel<<<N, 256, 0, stream>>>(a, nullptr, ALN, nullptr, CA);
  ln_kernel<<<N, 256, 0, stream>>>(s, gamma_s1, nullptr, SLN1, CS);
  ln_kernel<<<N, 256, 0, stream>>>(s, gamma_s2, nullptr, SLN2, CS);

  // --- pair bias: LN(z)*gamma_z @ Wz -> [H][N][N] (TDM-staged, WMMA) ---
  zbias_kernel<<<(N * N) / 64, 128, 0, stream>>>(z, gamma_z, WzT, BIAS);

  // --- adaLN #1 ---
  gemm(SLN1, CS, Wg1T, CS, bg1, G1, CA, N, CA, CS);
  gemm(SLN1, CS, Wa1T, CS, nullptr, ADD1, CA, N, CA, CS);
  ew_adaln_kernel<<<(N * CA + 255) / 256, 256, 0, stream>>>(G1, ADD1, ALN, ALN1b, N * CA);

  // --- QKV + gate projections ---
  gemm(ALN1b, CA, WqT, CA, bq, Qf, CA, N, CA, CA);
  gemm(ALN1b, CA, WkT, CA, nullptr, Kf, CA, N, CA, CA);
  gemm(ALN1b, CA, WvT, CA, nullptr, Vf, CA, N, CA, CA);
  gemm(ALN1b, CA, WgT, CA, bgate, GATEf, CA, N, CA, CA);
  pack_qk_kernel<<<(H * N * DP + 255) / 256, 256, 0, stream>>>(Qf, Qb);
  pack_qk_kernel<<<(H * N * DP + 255) / 256, 256, 0, stream>>>(Kf, Kb);
  pack_vt_kernel<<<(H * 48 * N + 255) / 256, 256, 0, stream>>>(Vf, VbT);

  // --- attention ---
  { dim3 g(N / 64, N / 16, H); scores_kernel<<<g, 32, 0, stream>>>(Qb, Kb, BIAS, SCORE); }
  softmax_kernel<<<H * N, 256, 0, stream>>>(SCORE);
  { dim3 g(1, N / 16, H); pv_kernel<<<g, 32, 0, stream>>>(SCORE, VbT, OH); }

  // --- output projection + gates ---
  ew_gatemul_kernel<<<(N * CA + 255) / 256, 256, 0, stream>>>(GATEf, OH, OGb, N * CA);
  gemm(OGb, CA, WoT, CA, bo, OOUT, CA, N, CA, CA);
  gemm(sb, CS, WslT, CS, bs_last, SLAST, CA, N, CA, CS);

  // --- conditioned transition block ---
  gemm(SLN2, CS, Wg2T, CS, bg2, G1, CA, N, CA, CS);        // reuse G1
  gemm(SLN2, CS, Wa2T, CS, nullptr, ADD1, CA, N, CA, CS);  // reuse ADD1
  ew_adaln_kernel<<<(N * CA + 255) / 256, 256, 0, stream>>>(G1, ADD1, ALN, A2b, N * CA);
  gemm(A2b, CA, W1T, CA, nullptr, X1, 2 * CA, N, 2 * CA, CA);
  gemm(A2b, CA, W2T, CA, nullptr, X2, 2 * CA, N, 2 * CA, CA);
  ew_silumul_kernel<<<(N * 2 * CA + 255) / 256, 256, 0, stream>>>(X1, X2, BMIDb, N * 2 * CA);
  gemm(BMIDb, 2 * CA, WbT, 2 * CA, nullptr, TOUT, CA, N, CA, 2 * CA);
  gemm(sb, CS, WstT, CS, b_t, ST, CA, N, CA, CS);

  // --- final combine ---
  final_kernel<<<(N * CA + 255) / 256, 256, 0, stream>>>(SLAST, OOUT, ST, TOUT,
                                                         (float*)d_out, N * CA);
}
